// EdgeBlock_33122787787020
// MI455X (gfx1250) — compile-verified, matrix operand order checked by reading, not run on hardware
//
#include <hip/hip_runtime.h>

typedef __attribute__((ext_vector_type(16))) __bf16 v16bf;
typedef __attribute__((ext_vector_type(8)))  __bf16 v8bf;
typedef __attribute__((ext_vector_type(4)))  __bf16 v4bf;
typedef __attribute__((ext_vector_type(8)))  float  v8f;
typedef __attribute__((ext_vector_type(4)))  float  v4f;

#define NODE_D   32
#define EDGE_D   16
#define GRAPH_D  16
#define IN_D     96
#define HID      32
#define OUT_D    32
#define BN_EPS   1e-5f

#define XT_STRIDE 120   // halves; 240B row stride: 16B aligned, conflict-free residues
#define HT_STRIDE 40    // halves; 80B row stride
#define OT_STRIDE 36    // floats; 144B row stride: 16B aligned, conflict-free residues
#define WPB       8     // waves per block (256 threads, wave32)

__device__ __forceinline__ v8f wmma_bf16(v16bf a, v16bf b, v8f c) {
  return __builtin_amdgcn_wmma_f32_16x16x32_bf16(false, a, false, b, (short)0, c,
                                                 false, false);
}

__device__ __forceinline__ v16bf make_frag(const __bf16* p0, const __bf16* p1) {
  v8bf lo = *(const v8bf*)p0;
  v8bf hi = *(const v8bf*)p1;
  return __builtin_shufflevector(lo, hi, 0,1,2,3,4,5,6,7,8,9,10,11,12,13,14,15);
}

__device__ __forceinline__ v4bf cvt4(v4f v) {
  v4bf h;
  h[0] = (__bf16)v[0]; h[1] = (__bf16)v[1];
  h[2] = (__bf16)v[2]; h[3] = (__bf16)v[3];
  return h;
}

// STATS=true: accumulate per-channel sum/sumsq of relu(x@W1+b1)
// STATS=false: fused out = relu(x@W1+b1) @ W2' + b2'
template <bool STATS>
__global__ void edge_mlp_kernel(const float* __restrict__ src,
                                const float* __restrict__ dest,
                                const float* __restrict__ edge_attr,
                                const float* __restrict__ u,
                                const int*   __restrict__ batch,
                                const float* __restrict__ W1,
                                const float* __restrict__ b1,
                                const __bf16* __restrict__ W2p,
                                const float* __restrict__ b2p,
                                float* __restrict__ out,
                                float* __restrict__ gsum,
                                float* __restrict__ gsq,
                                int E, int nTiles) {
  __shared__ __align__(16) __bf16 xt_all[WPB][16][XT_STRIDE];
  __shared__ __align__(16) __bf16 ht_all[WPB][16][HT_STRIDE];
  __shared__ __align__(16) float  ot_all[WPB][16][OT_STRIDE];
  __shared__ int   bt_all[WPB][16];
  __shared__ float bsum[HID];
  __shared__ float bsq[HID];

  const int lane = threadIdx.x & 31;
  const int wave = threadIdx.x >> 5;
  __bf16 (*xt)[XT_STRIDE] = xt_all[wave];
  __bf16 (*ht)[HT_STRIDE] = ht_all[wave];
  float  (*ot)[OT_STRIDE] = ot_all[wave];
  int*    bt              = bt_all[wave];

  if (STATS) {
    if (threadIdx.x < HID) { bsum[threadIdx.x] = 0.f; bsq[threadIdx.x] = 0.f; }
    __syncthreads();
  }

  const int colN = lane & 15;        // N index of this lane's B/C slice
  const int g    = (lane >> 4) * 8;  // K-half / M-half selector

  // ---- persistent B fragments for W1 (f32 [96][32] row-major) ----
  v16bf bw1[2][3];
  for (int nc = 0; nc < 2; ++nc) {
    const int col = colN + nc * 16;
#pragma unroll
    for (int ks = 0; ks < 3; ++ks) {
      const int kb = ks * 32;
      v16bf f;
#pragma unroll
      for (int e = 0; e < 8; ++e) {
        f[e]     = (__bf16)W1[(kb + g + e) * HID + col];
        f[e + 8] = (__bf16)W1[(kb + 16 + g + e) * HID + col];
      }
      bw1[nc][ks] = f;
    }
  }
  const float b1v0 = b1[colN];
  const float b1v1 = b1[colN + 16];

  // ---- persistent B fragments for W2' (bf16 [32][32]) + b2' ----
  v16bf bw2[2];
  float b2v[2] = {0.f, 0.f};
  if (!STATS) {
    for (int nc = 0; nc < 2; ++nc) {
      const int col = colN + nc * 16;
      v16bf f;
#pragma unroll
      for (int e = 0; e < 8; ++e) {
        f[e]     = W2p[(g + e) * OUT_D + col];
        f[e + 8] = W2p[(16 + g + e) * OUT_D + col];
      }
      bw2[nc] = f;
      b2v[nc] = b2p[col];
    }
  }

  float accS0 = 0.f, accS1 = 0.f, accQ0 = 0.f, accQ1 = 0.f;

  const int waveGlobal = blockIdx.x * WPB + wave;
  const int waveCount  = gridDim.x * WPB;
  const int row = lane & 15;

  for (int tile = waveGlobal; tile < nTiles; tile += waveCount) {
    const int base = tile * 16;

    // ---- stage per-row batch ids ----
    if (lane < 16) {
      int e = base + lane; if (e >= E) e = E - 1;
      bt[lane] = batch[e];
    }
    __asm__ volatile("" ::: "memory");

    // ---- gather x tile [16 x 96] as 12 coalesced global_load_b128 per lane,
    //      in 4 segment phases (no per-iteration pointer selects) ----
#pragma unroll
    for (int i = 0; i < 4; ++i) {          // src: 128 float4 quads
      const int q = lane + 32 * i;
      const int r = q >> 3, c = q & 7;
      int e = base + r; if (e >= E) e = E - 1;
      const v4f v = *(const v4f*)(src + (size_t)e * NODE_D + (size_t)(c * 4));
      *(v4bf*)&xt[r][c * 4] = cvt4(v);
    }
#pragma unroll
    for (int i = 0; i < 4; ++i) {          // dest: 128 float4 quads
      const int q = lane + 32 * i;
      const int r = q >> 3, c = q & 7;
      int e = base + r; if (e >= E) e = E - 1;
      const v4f v = *(const v4f*)(dest + (size_t)e * NODE_D + (size_t)(c * 4));
      *(v4bf*)&xt[r][32 + c * 4] = cvt4(v);
    }
#pragma unroll
    for (int i = 0; i < 2; ++i) {          // edge_attr: 64 float4 quads
      const int q = lane + 32 * i;
      const int r = q >> 2, c = q & 3;
      int e = base + r; if (e >= E) e = E - 1;
      const v4f v = *(const v4f*)(edge_attr + (size_t)e * EDGE_D + (size_t)(c * 4));
      *(v4bf*)&xt[r][64 + c * 4] = cvt4(v);
    }
#pragma unroll
    for (int i = 0; i < 2; ++i) {          // u[batch]: 64 float4 quads
      const int q = lane + 32 * i;
      const int r = q >> 2, c = q & 3;
      const v4f v = *(const v4f*)(u + (size_t)bt[r] * GRAPH_D + (size_t)(c * 4));
      *(v4bf*)&xt[r][80 + c * 4] = cvt4(v);
    }
    __asm__ volatile("" ::: "memory");  // DS stores precede DS loads (in-order per wave)

    // ---- GEMM1: [16x96] @ [96x32]; preload all A frags, then 6 WMMAs ----
    v16bf a0 = make_frag(&xt[row][0  + g], &xt[row][16 + g]);
    v16bf a1 = make_frag(&xt[row][32 + g], &xt[row][48 + g]);
    v16bf a2 = make_frag(&xt[row][64 + g], &xt[row][80 + g]);
    v8f c0 = {}, c1 = {};
    c0 = wmma_bf16(a0, bw1[0][0], c0);
    c1 = wmma_bf16(a0, bw1[1][0], c1);
    c0 = wmma_bf16(a1, bw1[0][1], c0);
    c1 = wmma_bf16(a1, bw1[1][1], c1);
    c0 = wmma_bf16(a2, bw1[0][2], c0);
    c1 = wmma_bf16(a2, bw1[1][2], c1);

    // ---- bias + relu; either reduce stats or stage h for GEMM2 ----
#pragma unroll
    for (int r = 0; r < 8; ++r) {
      const int M = r + g;  // C/D layout: lanes 16-31 hold M = r + 8
      float h0 = c0[r] + b1v0; h0 = h0 > 0.f ? h0 : 0.f;
      float h1 = c1[r] + b1v1; h1 = h1 > 0.f ? h1 : 0.f;
      if (STATS) {
        accS0 += h0; accQ0 += h0 * h0;
        accS1 += h1; accQ1 += h1 * h1;
      } else {
        ht[M][colN]      = (__bf16)h0;
        ht[M][colN + 16] = (__bf16)h1;
      }
    }

    if (!STATS) {
      __asm__ volatile("" ::: "memory");
      // ---- GEMM2: [16x32] @ [32x32], one WMMA per N-tile ----
      v16bf af = make_frag(&ht[row][g], &ht[row][16 + g]);
      v8f d0 = {}, d1 = {};
      d0 = wmma_bf16(af, bw2[0], d0);
      d1 = wmma_bf16(af, bw2[1], d1);

      // stage result tile in LDS, then 4x coalesced global_store_b128 per lane
#pragma unroll
      for (int r = 0; r < 8; ++r) {
        const int M = r + g;
        ot[M][colN]      = d0[r] + b2v[0];
        ot[M][colN + 16] = d1[r] + b2v[1];
      }
      __asm__ volatile("" ::: "memory");
#pragma unroll
      for (int j = 0; j < 4; ++j) {
        const int q  = lane + 32 * j;   // [0,128) float4 index in tile
        const int r  = q >> 3;
        const int c4 = q & 7;
        const v4f v = *(const v4f*)&ot[r][c4 * 4];
        const int e = base + r;
        if (e < E) *(v4f*)&out[(size_t)e * OUT_D + (size_t)(c4 * 4)] = v;
      }
    }
  }

  if (STATS) {
    atomicAdd(&bsum[colN],      accS0);
    atomicAdd(&bsum[colN + 16], accS1);
    atomicAdd(&bsq[colN],       accQ0);
    atomicAdd(&bsq[colN + 16],  accQ1);
    __syncthreads();
    if (threadIdx.x < HID) {
      atomicAdd(&gsum[threadIdx.x], bsum[threadIdx.x]);
      atomicAdd(&gsq[threadIdx.x],  bsq[threadIdx.x]);
    }
  }
}

__global__ void init_ws_kernel(float* ws) {
  if (threadIdx.x < 64) ws[threadIdx.x] = 0.f;
}

// Fold BN into GEMM2: W2'[k][n] = s[k]*W2[k][n] (bf16), b2'[n] = b2[n] + sum_k t[k]*W2[k][n]
__global__ void prep_kernel(const float* __restrict__ gsum, const float* __restrict__ gsq,
                            const float* __restrict__ gamma, const float* __restrict__ beta,
                            const float* __restrict__ W2, const float* __restrict__ b2,
                            __bf16* __restrict__ W2p, float* __restrict__ b2p, float invE) {
  __shared__ float s_[HID];
  __shared__ float t_[HID];
  const int t = threadIdx.x;
  if (t < HID) {
    const float mean = gsum[t] * invE;
    const float var  = gsq[t] * invE - mean * mean;
    const float sc   = gamma[t] * rsqrtf(var + BN_EPS);
    s_[t] = sc;
    t_[t] = beta[t] - mean * sc;
  }
  __syncthreads();
  if (t < HID) {
    for (int n = 0; n < OUT_D; ++n)
      W2p[t * OUT_D + n] = (__bf16)(s_[t] * W2[t * OUT_D + n]);
    float acc = b2[t];
    for (int k = 0; k < HID; ++k) acc += t_[k] * W2[k * OUT_D + t];
    b2p[t] = acc;
  }
}

extern "C" void kernel_launch(void* const* d_in, const int* in_sizes, int n_in,
                              void* d_out, int out_size, void* d_ws, size_t ws_size,
                              hipStream_t stream) {
  (void)n_in; (void)out_size; (void)ws_size;
  const float* src       = (const float*)d_in[0];
  const float* dest      = (const float*)d_in[1];
  const float* edge_attr = (const float*)d_in[2];
  const float* u         = (const float*)d_in[3];
  const int*   batch     = (const int*)d_in[4];
  const float* W1        = (const float*)d_in[5];
  const float* b1        = (const float*)d_in[6];
  const float* gamma     = (const float*)d_in[7];
  const float* beta      = (const float*)d_in[8];
  const float* W2        = (const float*)d_in[9];
  const float* b2        = (const float*)d_in[10];
  float* out = (float*)d_out;

  const int E      = in_sizes[0] / NODE_D;
  const int nTiles = (E + 15) / 16;

  float*  ws   = (float*)d_ws;
  float*  gsum = ws;                  // [32]
  float*  gsq  = ws + 32;             // [32]
  float*  b2p  = ws + 64;             // [32]
  __bf16* W2p  = (__bf16*)(ws + 96);  // [32][32] bf16

  init_ws_kernel<<<1, 64, 0, stream>>>(ws);

  const int blocks = 2048;  // 16384 waves grid-striding 125k tiles
  edge_mlp_kernel<true><<<blocks, 256, 0, stream>>>(
      src, dest, edge_attr, u, batch, W1, b1,
      (const __bf16*)nullptr, (const float*)nullptr, (float*)nullptr,
      gsum, gsq, E, nTiles);

  prep_kernel<<<1, 32, 0, stream>>>(gsum, gsq, gamma, beta, W2, b2, W2p, b2p,
                                    1.0f / (float)E);

  edge_mlp_kernel<false><<<blocks, 256, 0, stream>>>(
      src, dest, edge_attr, u, batch, W1, b1, W2p, b2p, out,
      (float*)nullptr, (float*)nullptr, E, nTiles);
}
